// GCNGraphRegression_44770739094124
// MI455X (gfx1250) — matrix operand order repaired and checked by reading, not compiled, not run on hardware
//
#include <hip/hip_runtime.h>
#include <hip/hip_bf16.h>

typedef __attribute__((ext_vector_type(16))) __bf16 v16bf;
typedef __attribute__((ext_vector_type(8)))  __bf16 v8bf;
typedef __attribute__((ext_vector_type(8)))  float  v8f;

#define BN_EPS 1e-5f

// ---------------- elementwise / setup kernels ----------------

__global__ void fill_f32_k(float* __restrict__ p, float v, long long n) {
  long long i = blockIdx.x * (long long)blockDim.x + threadIdx.x;
  if (i < n) p[i] = v;
}

__global__ void deg_accum_k(float* __restrict__ deg, const int* __restrict__ dst, int E) {
  int i = blockIdx.x * blockDim.x + threadIdx.x;
  if (i < E) unsafeAtomicAdd(&deg[dst[i]], 1.0f);
}

__global__ void dinv_k(float* __restrict__ deg, int n) {
  int i = blockIdx.x * blockDim.x + threadIdx.x;
  if (i < n) deg[i] = rsqrtf(deg[i]);   // deg >= 1 always (self loop)
}

__global__ void norm_k(float* __restrict__ nrm, const float* __restrict__ dinv,
                       const int* __restrict__ src, const int* __restrict__ dst,
                       int E, int N) {
  int i = blockIdx.x * blockDim.x + threadIdx.x;
  if (i < E) {
    nrm[i] = dinv[src[i]] * dinv[dst[i]];
  } else if (i < E + N) {
    float d = dinv[i - E];
    nrm[i] = d * d;                     // self loop: src == dst
  }
}

__global__ void f32_to_bf16_k(const float* __restrict__ in, __bf16* __restrict__ out, long long n) {
  long long i = blockIdx.x * (long long)blockDim.x + threadIdx.x;
  if (i < n) out[i] = (__bf16)in[i];
}

// W [K, Nout] fp32 row-major  ->  Wt [Nout, K] bf16 (transposed)
__global__ void wconvert_k(const float* __restrict__ W, __bf16* __restrict__ Wt, int K, int Nout) {
  int i = blockIdx.x * blockDim.x + threadIdx.x;
  if (i < K * Nout) {
    int k = i / Nout, n = i % Nout;
    Wt[(long long)n * K + k] = (__bf16)W[i];
  }
}

// ---------------- WMMA bf16 GEMM: C[M,Nout] = A[M,K] * Bt[Nout,K]^T ----------------
// One wave computes a 16(M) x 128(N) strip: 8 accumulators, A-fragment reused 8x.
#define GEMM_NT 8

__global__ __launch_bounds__(256)
void gemm_bf16_k(const __bf16* __restrict__ A, const __bf16* __restrict__ Bt,
                 float* __restrict__ C, const float* __restrict__ bias,
                 int M, int K, int Nout, int relu) {
  int wave = (blockIdx.x * blockDim.x + threadIdx.x) >> 5;
  int lane = threadIdx.x & 31;
  int ntG  = Nout / (16 * GEMM_NT);     // groups of 128 columns
  int tm   = wave / ntG;
  int tg   = wave % ntG;
  if (tm >= (M >> 4)) return;           // wave-uniform guard (EXEC stays all-ones)

  int l15     = lane & 15;
  int halfSel = lane >> 4;              // 0: lanes 0-15, 1: lanes 16-31
  const __bf16* arow = A + (long long)(tm * 16 + l15) * K;
  int aoff = halfSel * 16;              // A K-offset: contiguous 16 K per lane half

  v8f acc[GEMM_NT] = {};

  for (int k0 = 0; k0 < K; k0 += 32) {
    // A fragment: elements 0..15 -> K = k0+aoff+e
    v16bf a = *(const v16bf*)(arow + k0 + aoff);

#pragma unroll
    for (int t = 0; t < GEMM_NT; ++t) {
      int col = (tg * GEMM_NT + t) * 16 + l15;
      // B fragment: 16 contiguous bf16 of column `col` at K = k0 + halfSel*16
      v16bf b = *(const v16bf*)(Bt + (long long)col * K + k0 + halfSel * 16);
      acc[t] = __builtin_amdgcn_wmma_f32_16x16x32_bf16(
          false, a, false, b, (short)0, acc[t], false, false);
    }
  }

  int rbase = tm * 16 + (halfSel ? 8 : 0);
#pragma unroll
  for (int t = 0; t < GEMM_NT; ++t) {
    int col = (tg * GEMM_NT + t) * 16 + l15;
    float bv = bias ? bias[col] : 0.0f;
#pragma unroll
    for (int r = 0; r < 8; ++r) {
      float v = acc[t][r] + bv;
      if (relu) v = fmaxf(v, 0.0f);
      C[(long long)(rbase + r) * Nout + col] = v;
    }
  }
}

// ---------------- message passing: hout[dst] += hin[src] * norm ----------------
// 128 threads per edge, each handles 4 consecutive features (float4).
__global__ void scatter_k(const float* __restrict__ hin, float* __restrict__ hout,
                          const int* __restrict__ src, const int* __restrict__ dst,
                          const float* __restrict__ nrm, int E, int nE, int Hdim) {
  long long t = blockIdx.x * (long long)blockDim.x + threadIdx.x;
  int perEdge = Hdim >> 2;
  long long e = t / perEdge;
  if (e >= nE) return;
  int j = (int)(t % perEdge) << 2;
  int s, d;
  if (e < E) { s = src[e]; d = dst[e]; }
  else       { s = d = (int)(e - E); }
  float w = nrm[e];
  float4 v = *(const float4*)(hin + (long long)s * Hdim + j);
  float* o = hout + (long long)d * Hdim + j;
  unsafeAtomicAdd(o + 0, v.x * w);
  unsafeAtomicAdd(o + 1, v.y * w);
  unsafeAtomicAdd(o + 2, v.z * w);
  unsafeAtomicAdd(o + 3, v.w * w);
}

// ---------------- BN: column stats (sum, sumsq) then normalize ----------------
__global__ void stats_k(const float* __restrict__ h, float* __restrict__ stats,
                        int Nrows, int Hdim, int rowsPerBlk) {
  int colBlocks = Hdim >> 8;            // Hdim / 256
  int cb = blockIdx.x % colBlocks;
  int rb = blockIdx.x / colBlocks;
  int col = cb * 256 + threadIdx.x;
  int r0 = rb * rowsPerBlk;
  int r1 = min(r0 + rowsPerBlk, Nrows);
  float s = 0.0f, s2 = 0.0f;
  for (int r = r0; r < r1; ++r) {
    float v = h[(long long)r * Hdim + col];
    s += v; s2 += v * v;
  }
  unsafeAtomicAdd(&stats[col], s);
  unsafeAtomicAdd(&stats[Hdim + col], s2);
}

// If out_bf != nullptr: write bf16 only (next layer's GEMM input, saves fp32 traffic).
// Else: write fp32 in place.
__global__ void bn_apply_k(float* __restrict__ h, __bf16* __restrict__ out_bf,
                           const float* __restrict__ stats,
                           const float* __restrict__ g, const float* __restrict__ beta,
                           int Nrows, int Hdim, int relu) {
  long long i = blockIdx.x * (long long)blockDim.x + threadIdx.x;
  if (i >= (long long)Nrows * Hdim) return;
  int col = (int)(i % Hdim);
  float inv_n = 1.0f / (float)Nrows;
  float mu  = stats[col] * inv_n;
  float var = stats[Hdim + col] * inv_n - mu * mu;
  float v = (h[i] - mu) * rsqrtf(var + BN_EPS) * g[col] + beta[col];
  if (relu) v = fmaxf(v, 0.0f);
  if (out_bf) out_bf[i] = (__bf16)v;
  else        h[i] = v;
}

// ---------------- global mean pool ----------------
__global__ void cnt_k(float* __restrict__ cnt, const int* __restrict__ bidx, int n) {
  int i = blockIdx.x * blockDim.x + threadIdx.x;
  if (i < n) unsafeAtomicAdd(&cnt[bidx[i]], 1.0f);
}

__global__ void pool_accum_k(const float* __restrict__ h, float* __restrict__ pooled,
                             const int* __restrict__ bidx, int n, int Hdim) {
  long long t = blockIdx.x * (long long)blockDim.x + threadIdx.x;
  int perNode = Hdim >> 2;
  long long node = t / perNode;
  if (node >= n) return;
  int j = (int)(t % perNode) << 2;
  int b = bidx[node];
  float4 v = *(const float4*)(h + node * Hdim + j);
  float* o = pooled + (long long)b * Hdim + j;
  unsafeAtomicAdd(o + 0, v.x);
  unsafeAtomicAdd(o + 1, v.y);
  unsafeAtomicAdd(o + 2, v.z);
  unsafeAtomicAdd(o + 3, v.w);
}

__global__ void pool_div_k(float* __restrict__ pooled, const float* __restrict__ cnt,
                           int G, int Hdim) {
  long long i = blockIdx.x * (long long)blockDim.x + threadIdx.x;
  if (i >= (long long)G * Hdim) return;
  pooled[i] /= fmaxf(cnt[i / Hdim], 1.0f);
}

// ---------------- lin2: out[g] = dot(p[g,:], w) + b ----------------
__global__ void lin2_k(const float* __restrict__ p, const float* __restrict__ w,
                       const float* __restrict__ b, float* __restrict__ out,
                       int G, int K) {
  int wave = (blockIdx.x * blockDim.x + threadIdx.x) >> 5;
  int lane = threadIdx.x & 31;
  if (wave >= G) return;
  float s = 0.0f;
  for (int k = lane; k < K; k += 32) s += p[(long long)wave * K + k] * w[k];
#pragma unroll
  for (int off = 16; off > 0; off >>= 1) s += __shfl_down(s, off, 32);
  if (lane == 0) out[wave] = s + b[0];
}

// ---------------- host orchestration ----------------

static inline int ceil_div(long long a, long long b) { return (int)((a + b - 1) / b); }

extern "C" void kernel_launch(void* const* d_in, const int* in_sizes, int n_in,
                              void* d_out, int out_size, void* d_ws, size_t ws_size,
                              hipStream_t stream) {
  const int N = 50000, G = 2048, D_IN = 128, H = 512, L = 4;

  const float* x      = (const float*)d_in[0];
  const int*   ei     = (const int*)d_in[1];
  const int    E      = in_sizes[1] / 2;
  const int*   src    = ei;
  const int*   dst    = ei + E;
  const int*   bidx   = (const int*)d_in[2];
  const float* W1     = (const float*)d_in[3];
  const float* Wh     = (const float*)d_in[5];   // [L-1, H, H]
  const float* bn_g   = (const float*)d_in[7];   // [L+1, H]
  const float* bn_b   = (const float*)d_in[8];
  const float* lin1W  = (const float*)d_in[9];
  const float* lin1b  = (const float*)d_in[10];
  const float* lin2W  = (const float*)d_in[11];
  const float* lin2b  = (const float*)d_in[12];
  float* out = (float*)d_out;

  char* ws = (char*)d_ws;
  size_t off = 0;
  auto carve = [&](size_t bytes) -> char* {
    char* p = ws + off;
    off += (bytes + 255) & ~(size_t)255;
    return p;
  };
  float*  deg    = (float*)carve((size_t)N * 4);
  float*  nrm    = (float*)carve((size_t)(E + N) * 4);
  float*  hG     = (float*)carve((size_t)N * H * 4);   // GEMM output
  float*  hS     = (float*)carve((size_t)N * H * 4);   // scatter output / activation
  __bf16* hbf    = (__bf16*)carve((size_t)N * H * 2);  // bf16 GEMM input
  __bf16* Wt     = (__bf16*)carve((size_t)H * H * 2);
  float*  stats  = (float*)carve((size_t)2 * H * 4);
  float*  pooled = (float*)carve((size_t)G * H * 4);
  float*  cnt    = (float*)carve((size_t)G * 4);
  __bf16* pbf    = (__bf16*)carve((size_t)G * H * 2);
  float*  p2     = (float*)carve((size_t)G * H * 4);

  const int B = 256;
  const int nE = E + N;

  // --- GCN normalization: deg (with self loop) -> deg^-1/2 -> edge norms
  fill_f32_k<<<ceil_div(N, B), B, 0, stream>>>(deg, 1.0f, N);
  deg_accum_k<<<ceil_div(E, B), B, 0, stream>>>(deg, dst, E);
  dinv_k<<<ceil_div(N, B), B, 0, stream>>>(deg, N);
  norm_k<<<ceil_div(nE, B), B, 0, stream>>>(nrm, deg, src, dst, E, N);

  // layer 0 GEMM input: x -> bf16 (later layers get bf16 straight from BN epilogue)
  f32_to_bf16_k<<<ceil_div((long long)N * D_IN, B), B, 0, stream>>>(x, hbf, (long long)N * D_IN);

  // --- L conv layers
  for (int l = 0; l < L; ++l) {
    int K = (l == 0) ? D_IN : H;
    const float* Wl = (l == 0) ? W1 : (Wh + (size_t)(l - 1) * H * H);

    // weight -> transposed bf16
    wconvert_k<<<ceil_div((long long)K * H, B), B, 0, stream>>>(Wl, Wt, K, H);

    // hG = act @ W  (WMMA bf16)
    {
      long long waves = (long long)(N / 16) * (H / (16 * GEMM_NT));
      gemm_bf16_k<<<ceil_div(waves * 32, B), B, 0, stream>>>(hbf, Wt, hG, nullptr, N, K, H, 0);
    }

    // hS = segment_sum over edges of hG[src]*norm
    fill_f32_k<<<ceil_div((long long)N * H, B), B, 0, stream>>>(hS, 0.0f, (long long)N * H);
    scatter_k<<<ceil_div((long long)nE * (H / 4), B), B, 0, stream>>>(hG, hS, src, dst, nrm, E, nE, H);

    // BatchNorm + ReLU (bias is absorbed by mean subtraction).
    // Layers 0..L-2 write bf16 directly (next GEMM input); last layer keeps fp32 for pooling.
    fill_f32_k<<<ceil_div(2 * H, B), B, 0, stream>>>(stats, 0.0f, 2 * H);
    {
      int rowsPerBlk = 256;
      int grid = ceil_div(N, rowsPerBlk) * (H / 256);
      stats_k<<<grid, B, 0, stream>>>(hS, stats, N, H, rowsPerBlk);
    }
    __bf16* bnOut = (l < L - 1) ? hbf : nullptr;
    bn_apply_k<<<ceil_div((long long)N * H, B), B, 0, stream>>>(
        hS, bnOut, stats, bn_g + (size_t)l * H, bn_b + (size_t)l * H, N, H, 1);
  }

  // --- global mean pool (from fp32 hS of last layer)
  fill_f32_k<<<ceil_div((long long)G * H, B), B, 0, stream>>>(pooled, 0.0f, (long long)G * H);
  fill_f32_k<<<ceil_div(G, B), B, 0, stream>>>(cnt, 0.0f, G);
  cnt_k<<<ceil_div(N, B), B, 0, stream>>>(cnt, bidx, N);
  pool_accum_k<<<ceil_div((long long)N * (H / 4), B), B, 0, stream>>>(hS, pooled, bidx, N, H);
  pool_div_k<<<ceil_div((long long)G * H, B), B, 0, stream>>>(pooled, cnt, G, H);

  // --- BN over pooled -> bf16 (lin1 GEMM input), no ReLU
  fill_f32_k<<<ceil_div(2 * H, B), B, 0, stream>>>(stats, 0.0f, 2 * H);
  {
    int rowsPerBlk = 128;
    int grid = ceil_div(G, rowsPerBlk) * (H / 256);
    stats_k<<<grid, B, 0, stream>>>(pooled, stats, G, H, rowsPerBlk);
  }
  bn_apply_k<<<ceil_div((long long)G * H, B), B, 0, stream>>>(
      pooled, pbf, stats, bn_g + (size_t)L * H, bn_b + (size_t)L * H, G, H, 0);

  // --- lin1 (+bias+ReLU fused) via WMMA
  wconvert_k<<<ceil_div((long long)H * H, B), B, 0, stream>>>(lin1W, Wt, H, H);
  {
    long long waves = (long long)(G / 16) * (H / (16 * GEMM_NT));
    gemm_bf16_k<<<ceil_div(waves * 32, B), B, 0, stream>>>(pbf, Wt, p2, lin1b, G, H, H, 1);
  }

  // --- lin2: [G,H] @ [H,1] + b
  lin2_k<<<ceil_div((long long)G * 32, B), B, 0, stream>>>(p2, lin2W, lin2b, out, G, H);
}
